// SpatialAttention_37409165148451
// MI455X (gfx1250) — matrix operand order repaired and checked by reading, not compiled, not run on hardware
//
#include <hip/hip_runtime.h>

typedef __bf16 bf16;
typedef __attribute__((ext_vector_type(16))) __bf16 v16bf;
typedef __attribute__((ext_vector_type(8)))  __bf16 v8bf;
typedef __attribute__((ext_vector_type(8)))  float   v8f;

#define D_MODEL   1024
#define NUM_HEADS 16
#define HEAD_DIM  64
#define BT        64
#define SEQ       256
#define M_TOTAL   (BT * SEQ)       // 16384
#define N_QKV     (3 * D_MODEL)    // 3072

#define AS1 __attribute__((address_space(1)))
#define AS3 __attribute__((address_space(3)))

#if __has_builtin(__builtin_amdgcn_tensor_load_to_lds)
#define HAVE_TDM 1
#else
#define HAVE_TDM 0
#endif
#if __has_builtin(__builtin_amdgcn_global_load_async_to_lds_b128)
#define HAVE_ASYNC 1
#else
#define HAVE_ASYNC 0
#endif

typedef __attribute__((ext_vector_type(4))) unsigned int v4u;
typedef __attribute__((ext_vector_type(8))) int          v8i_;
typedef __attribute__((ext_vector_type(4))) int          v4i_;

// ---------- fragment helpers (layouts per CDNA5 ISA 7.12.2) ----------
static __device__ __forceinline__ v16bf ld_frag16(const bf16* lo, const bf16* hi) {
  union { v16bf v; v8bf h[2]; } u;
  u.h[0] = *(const v8bf*)lo;
  u.h[1] = *(const v8bf*)hi;
  return u.v;
}

static __device__ __forceinline__ v8f wmma_bf16(v16bf a, v16bf b, v8f c) {
  // D = A(16x32 bf16) * B(32x16 bf16) + C(16x16 f32)
  return __builtin_amdgcn_wmma_f32_16x16x32_bf16(false, a, false, b, (short)0, c,
                                                 false, false);
}

// ---------- Tensor Data Mover: 2-D tile Global -> LDS (CDNA5 ISA ch.8) ----------
#if HAVE_TDM
static __device__ __forceinline__ void tdm_load_2d(const bf16* gsrc, const bf16* ldst,
                                                   unsigned tile_w, unsigned tile_h,
                                                   unsigned tensor_w, unsigned tensor_h,
                                                   unsigned row_stride /*elements*/) {
  unsigned lds_addr = (unsigned)(size_t)(AS3 const bf16*)ldst;
  unsigned long long ga = (unsigned long long)(size_t)gsrc;
  // D# group 0: count=1 [1:0], lds_addr [63:32], global_addr [120:64], type=2 [127:126]
  union { unsigned long long q[2]; v4u v; } g0;
  g0.q[0] = 1ull | ((unsigned long long)lds_addr << 32);
  g0.q[1] = (ga & 0x01FFFFFFFFFFFFFFull) | (2ull << 62);
  // D# group 1: data_size=1(2B) [17:16], tensor_dim0 [79:48], tensor_dim1 [111:80],
  //             tile_dim0 [127:112], tile_dim1 [143:128], dim0_stride [207:160]
  union { unsigned long long q[4]; v8i_ v; } g1;
  g1.q[0] = (1ull << 16) | ((unsigned long long)(tensor_w & 0xFFFFu) << 48);
  g1.q[1] = (unsigned long long)(tensor_w >> 16)
          | ((unsigned long long)tensor_h << 16)
          | ((unsigned long long)(tile_w & 0xFFFFu) << 48);
  g1.q[2] = (unsigned long long)(tile_h & 0xFFFFu)
          | ((unsigned long long)row_stride << 32);
  g1.q[3] = 0ull;  // dim0_stride[47:32]=0, dim1_stride=0 (2-D tile)
  v4i_ z4 = {0, 0, 0, 0};                    // groups 2/3 unused for 2-D tiles
  v8i_ z8 = {0, 0, 0, 0, 0, 0, 0, 0};
  __builtin_amdgcn_tensor_load_to_lds(g0.v, g1.v, z4, z4, z8, 0);
}
#endif

#if HAVE_ASYNC
static __device__ __forceinline__ void async_cp16(const bf16* gsrc, bf16* ldst) {
  __builtin_amdgcn_global_load_async_to_lds_b128((AS1 v4i_*)gsrc, (AS3 v4i_*)ldst, 0, 0);
}
static __device__ __forceinline__ void wait_async0() {
#if __has_builtin(__builtin_amdgcn_s_wait_asynccnt)
  __builtin_amdgcn_s_wait_asynccnt(0);
#elif defined(__gfx1250__)
  asm volatile("s_wait_asynccnt 0x0" ::: "memory");
#endif
}
#endif

// ---------- 0) fp32 -> bf16 conversion ----------
__global__ __launch_bounds__(256) void f32_to_bf16(const float* __restrict__ in,
                                                   bf16* __restrict__ out, int n) {
  int i = blockIdx.x * 256 + threadIdx.x;
  if (i < n) out[i] = (bf16)in[i];
}

// ---------- 1/4) C[M,N] = A[M,K] * B[N,K]^T  (bf16 in, fp32 acc) ----------
// Block tile 128x128, 4 waves (2x2) of 64x64, K-step 64.
// TDM path: double-buffered LDS, DMA of tile i+1 overlaps compute of tile i.
template <bool OUT_F32>
__global__ __launch_bounds__(128) void gemm_bf16_nt(const bf16* __restrict__ A,
                                                    const bf16* __restrict__ B,
                                                    bf16* __restrict__ Cb,
                                                    float* __restrict__ Cf,
                                                    const float* __restrict__ bias,
                                                    int M, int N, int K) {
  __shared__ bf16 As[2][128 * 64];
  __shared__ bf16 Bs[2][128 * 64];

  const int m0   = blockIdx.y * 128;
  const int n0   = blockIdx.x * 128;
  const int tid  = threadIdx.x;
  const int wave = tid >> 5;
  const int lane = tid & 31;
  const int wm   = (wave >> 1) * 64;
  const int wn   = (wave & 1) * 64;
  const int halo = lane >> 4;     // 0: lanes 0-15, 1: lanes 16-31
  const int lan  = lane & 15;

  v8f acc[4][4] = {};
  const int nk = K >> 6;

#if HAVE_TDM
  if (wave == 0) {   // prologue: fill buffer 0
    tdm_load_2d(&A[(size_t)m0 * K], As[0], 64, 128, (unsigned)K, (unsigned)M, (unsigned)K);
    tdm_load_2d(&B[(size_t)n0 * K], Bs[0], 64, 128, (unsigned)K, (unsigned)N, (unsigned)K);
  }
#endif

  for (int it = 0; it < nk; ++it) {
    const int b  = it & 1;
    const int kt = it << 6;
#if HAVE_TDM
    if (wave == 0) {
      if (it + 1 < nk) {   // prefetch next tile into alternate buffer
        tdm_load_2d(&A[(size_t)m0 * K + kt + 64], As[b ^ 1], 64, 128,
                    (unsigned)K, (unsigned)M, (unsigned)K);
        tdm_load_2d(&B[(size_t)n0 * K + kt + 64], Bs[b ^ 1], 64, 128,
                    (unsigned)K, (unsigned)N, (unsigned)K);
        __builtin_amdgcn_s_wait_tensorcnt(2);  // current tile resident; next in flight
      } else {
        __builtin_amdgcn_s_wait_tensorcnt(0);
      }
    }
    const bf16* Asb = As[b];
    const bf16* Bsb = Bs[b];
#elif HAVE_ASYNC
#pragma unroll
    for (int i = 0; i < 8; ++i) {
      int c   = tid + (i << 7);
      int row = c >> 3;
      int col = (c & 7) * 8;
      async_cp16(&A[(size_t)(m0 + row) * K + kt + col], &As[0][row * 64 + col]);
      async_cp16(&B[(size_t)(n0 + row) * K + kt + col], &Bs[0][row * 64 + col]);
    }
    wait_async0();
    const bf16* Asb = As[0];
    const bf16* Bsb = Bs[0];
#else
#pragma unroll
    for (int i = 0; i < 8; ++i) {
      int c   = tid + (i << 7);
      int row = c >> 3;
      int col = (c & 7) * 8;
      *(v8bf*)&As[0][row * 64 + col] = *(const v8bf*)&A[(size_t)(m0 + row) * K + kt + col];
      *(v8bf*)&Bs[0][row * 64 + col] = *(const v8bf*)&B[(size_t)(n0 + row) * K + kt + col];
    }
    const bf16* Asb = As[0];
    const bf16* Bsb = Bs[0];
#endif
    __syncthreads();

#pragma unroll
    for (int kc = 0; kc < 2; ++kc) {
      const int kb = kc * 32;
      v16bf af[4], bfg[4];
#pragma unroll
      for (int mf = 0; mf < 4; ++mf) {
        const bf16* p = &Asb[(wm + mf * 16 + lan) * 64 + kb + halo * 8];
        af[mf] = ld_frag16(p, p + 16);   // K {0..7,16..23} / {8..15,24..31}
      }
#pragma unroll
      for (int nf = 0; nf < 4; ++nf) {
        const bf16* p = &Bsb[(wn + nf * 16 + lan) * 64 + kb + halo * 8];
        bfg[nf] = ld_frag16(p, p + 16);   // K {0..7,16..23} / {8..15,24..31}
      }
#pragma unroll
      for (int mf = 0; mf < 4; ++mf)
#pragma unroll
        for (int nf = 0; nf < 4; ++nf)
          acc[mf][nf] = wmma_bf16(af[mf], bfg[nf], acc[mf][nf]);
    }
    __syncthreads();
  }

  // C layout: vgpr r -> row = r + 8*halo, col = lane&15
#pragma unroll
  for (int mf = 0; mf < 4; ++mf)
#pragma unroll
    for (int nf = 0; nf < 4; ++nf)
#pragma unroll
      for (int r = 0; r < 8; ++r) {
        int row = m0 + wm + mf * 16 + r + halo * 8;
        int col = n0 + wn + nf * 16 + lan;
        if (OUT_F32)
          Cf[(size_t)row * N + col] = acc[mf][nf][r] + bias[col];
        else
          Cb[(size_t)row * N + col] = (bf16)acc[mf][nf][r];
      }
}

// ---------- 2) 2D RoPE in place on q,k halves of qkv (bf16) ----------
// Also folds the attention scale 1/sqrt(64) into q.
__global__ __launch_bounds__(256) void rope2d(bf16* __restrict__ qkv) {
  int id = blockIdx.x * 256 + threadIdx.x;     // 16384*1024 pair-threads
  int m    = id >> 10;
  int r    = id & 1023;
  int qk   = r >> 9;          // 0 = q, 1 = k
  int r2   = r & 511;
  int head = r2 >> 5;
  int p    = r2 & 31;
  int axis = p >> 4;          // 0: row(h) pos, 1: col(w) pos
  int j    = p & 15;
  int s    = m & 255;
  int pos  = axis ? (s & 15) : (s >> 4);

  // inv_freq = 10000^(-2j/32) = exp(-j * ln(10000)/16)
  float inv = __expf(-0.5756462732485114f * (float)j);
  float ang = (float)pos * inv;
  float cs, sn;
  __sincosf(ang, &sn, &cs);
  float sc = qk ? 1.0f : 0.125f;   // fold 1/sqrt(HEAD_DIM) into q

  size_t base = (size_t)m * N_QKV + qk * D_MODEL + head * HEAD_DIM + axis * 32 + j;
  float t1 = (float)qkv[base];
  float t2 = (float)qkv[base + 16];
  qkv[base]      = (bf16)((t1 * cs - t2 * sn) * sc);
  qkv[base + 16] = (bf16)((t2 * cs + t1 * sn) * sc);
}

// ---------- 3) attention: one (bt, head) per 256-thread block ----------
#define VSTRIDE 272   // padded transposed-V row stride (16B aligned)
__global__ __launch_bounds__(256) void attn_kernel(const bf16* __restrict__ qkv,
                                                   bf16* __restrict__ o) {
  __shared__ bf16 Ks[SEQ * 64];              // K tile, row-major [j][d]
  __shared__ bf16 Vt[64 * VSTRIDE];          // V transposed [d][j]
  __shared__ bf16 Ps[8][32 * 32];            // per-wave P scratch (C->A relayout)

  const int bt    = blockIdx.x >> 4;
  const int head  = blockIdx.x & 15;
  const int tid   = threadIdx.x;
  const int wave  = tid >> 5;
  const int lane  = tid & 31;
  const int halo  = lane >> 4;
  const int lan   = lane & 15;
  const size_t rbase = (size_t)bt * SEQ;
  const int kcol0 = D_MODEL + head * HEAD_DIM;
  const int vcol0 = 2 * D_MODEL + head * HEAD_DIM;

  // stage K: one TDM descriptor (256x64 tile, row stride 3072)
#if HAVE_TDM
  if (wave == 0)
    tdm_load_2d(&qkv[rbase * N_QKV + kcol0], Ks, 64, SEQ, N_QKV, M_TOTAL, N_QKV);
#else
#pragma unroll
  for (int i = 0; i < 8; ++i) {
    int c   = tid + (i << 8);
    int row = c >> 3;
    int col = (c & 7) * 8;
    *(v8bf*)&Ks[row * 64 + col] =
        *(const v8bf*)&qkv[(rbase + row) * N_QKV + kcol0 + col];
  }
#endif

  // stage V transposed (needs per-element scatter; through VGPRs)
#pragma unroll
  for (int i = 0; i < 8; ++i) {
    int c   = tid + (i << 8);
    int row = c >> 3;
    int col = (c & 7) * 8;
    v8bf vv = *(const v8bf*)&qkv[(rbase + row) * N_QKV + vcol0 + col];
#pragma unroll
    for (int e = 0; e < 8; ++e) Vt[(col + e) * VSTRIDE + row] = vv[e];
  }
#if HAVE_TDM
  if (wave == 0) __builtin_amdgcn_s_wait_tensorcnt(0);
#endif
  __syncthreads();

  // per-wave Q fragments (32 rows), loaded straight from global (scale pre-folded)
  const int qr = wave * 32;
  v16bf qa[2][2];
#pragma unroll
  for (int mf = 0; mf < 2; ++mf)
#pragma unroll
    for (int dc = 0; dc < 2; ++dc) {
      const bf16* p = &qkv[(rbase + qr + mf * 16 + lan) * N_QKV +
                           head * HEAD_DIM + dc * 32 + halo * 8];
      qa[mf][dc] = ld_frag16(p, p + 16);
    }

  float mrow[2][8], lrow[2][8];
  v8f oacc[2][4] = {};
#pragma unroll
  for (int mf = 0; mf < 2; ++mf)
#pragma unroll
    for (int r = 0; r < 8; ++r) { mrow[mf][r] = -1e30f; lrow[mf][r] = 0.f; }

  for (int j0 = 0; j0 < SEQ; j0 += 32) {
    // S chunk 32x32 = Q (32x64) * K^T
    v16bf kb[2][2];
#pragma unroll
    for (int nf = 0; nf < 2; ++nf)
#pragma unroll
      for (int dc = 0; dc < 2; ++dc) {
        const bf16* p = &Ks[(j0 + nf * 16 + lan) * 64 + dc * 32 + halo * 8];
        kb[nf][dc] = ld_frag16(p, p + 16);
      }
    v8f s[2][2] = {};
#pragma unroll
    for (int mf = 0; mf < 2; ++mf)
#pragma unroll
      for (int nf = 0; nf < 2; ++nf)
#pragma unroll
        for (int dc = 0; dc < 2; ++dc)
          s[mf][nf] = wmma_bf16(qa[mf][dc], kb[nf][dc], s[mf][nf]);

    // online softmax per row; rows live across 16 lanes of a half-wave
#pragma unroll
    for (int mf = 0; mf < 2; ++mf)
#pragma unroll
      for (int r = 0; r < 8; ++r) {
        float a0 = s[mf][0][r];
        float a1 = s[mf][1][r];
        float mx = fmaxf(a0, a1);
        mx = fmaxf(mx, __shfl_xor(mx, 1));
        mx = fmaxf(mx, __shfl_xor(mx, 2));
        mx = fmaxf(mx, __shfl_xor(mx, 4));
        mx = fmaxf(mx, __shfl_xor(mx, 8));
        float mnew  = fmaxf(mrow[mf][r], mx);
        float alpha = __expf(mrow[mf][r] - mnew);
        float p0 = __expf(a0 - mnew);
        float p1 = __expf(a1 - mnew);
        float ps = p0 + p1;
        ps += __shfl_xor(ps, 1);
        ps += __shfl_xor(ps, 2);
        ps += __shfl_xor(ps, 4);
        ps += __shfl_xor(ps, 8);
        lrow[mf][r] = lrow[mf][r] * alpha + ps;
        mrow[mf][r] = mnew;
#pragma unroll
        for (int nf = 0; nf < 4; ++nf) oacc[mf][nf][r] *= alpha;
        int prow = mf * 16 + r + halo * 8;
        Ps[wave][prow * 32 + lan]      = (bf16)p0;
        Ps[wave][prow * 32 + 16 + lan] = (bf16)p1;
      }

    // O += P (32x32) * V (32x64); P reloaded as A-frags from private LDS patch
    v16bf pa[2];
#pragma unroll
    for (int mf = 0; mf < 2; ++mf) {
      const bf16* p = &Ps[wave][(mf * 16 + lan) * 32 + halo * 8];
      pa[mf] = ld_frag16(p, p + 16);
    }
#pragma unroll
    for (int nf = 0; nf < 4; ++nf) {
      const bf16* p = &Vt[(nf * 16 + lan) * VSTRIDE + j0 + halo * 8];
      v16bf vb = ld_frag16(p, p + 16);
#pragma unroll
      for (int mf = 0; mf < 2; ++mf)
        oacc[mf][nf] = wmma_bf16(pa[mf], vb, oacc[mf][nf]);
    }
  }

  // normalize and store O (bf16, [m, head*64 + d])
#pragma unroll
  for (int mf = 0; mf < 2; ++mf)
#pragma unroll
    for (int nf = 0; nf < 4; ++nf)
#pragma unroll
      for (int r = 0; r < 8; ++r) {
        size_t row = rbase + qr + mf * 16 + r + halo * 8;
        int col = head * HEAD_DIM + nf * 16 + lan;
        o[row * D_MODEL + col] = (bf16)(oacc[mf][nf][r] / lrow[mf][r]);
      }
}

// ---------- host-side launcher ----------
extern "C" void kernel_launch(void* const* d_in, const int* in_sizes, int n_in,
                              void* d_out, int out_size, void* d_ws, size_t ws_size,
                              hipStream_t stream) {
  const float* x      = (const float*)d_in[0];
  const float* w_qkv  = (const float*)d_in[1];
  const float* w_proj = (const float*)d_in[2];
  const float* b_proj = (const float*)d_in[3];
  float* out = (float*)d_out;

  char* ws = (char*)d_ws;
  bf16* xb     = (bf16*)(ws + 0);                       // 16384*1024  = 32 MiB
  bf16* wqkvb  = (bf16*)(ws + (size_t)33554432);        //  3072*1024  =  6 MiB
  bf16* wprojb = (bf16*)(ws + (size_t)39845888);        //  1024*1024  =  2 MiB
  bf16* qkvb   = (bf16*)(ws + (size_t)41943040);        // 16384*3072  = 96 MiB
  bf16* ob     = (bf16*)(ws + (size_t)142606336);       // 16384*1024  = 32 MiB

  const int n_x  = M_TOTAL * D_MODEL;       // 16777216
  const int n_wq = N_QKV * D_MODEL;         // 3145728
  const int n_wp = D_MODEL * D_MODEL;       // 1048576

  f32_to_bf16<<<(n_x  + 255) / 256, 256, 0, stream>>>(x,      xb,     n_x);
  f32_to_bf16<<<(n_wq + 255) / 256, 256, 0, stream>>>(w_qkv,  wqkvb,  n_wq);
  f32_to_bf16<<<(n_wp + 255) / 256, 256, 0, stream>>>(w_proj, wprojb, n_wp);

  // qkv = x @ w_qkv^T   (M=16384, N=3072, K=1024)
  gemm_bf16_nt<false><<<dim3(N_QKV / 128, M_TOTAL / 128), 128, 0, stream>>>(
      xb, wqkvb, qkvb, nullptr, nullptr, M_TOTAL, N_QKV, D_MODEL);

  // 2D RoPE on q and k halves in place (+ q pre-scaled by 1/sqrt(dh))
  rope2d<<<(M_TOTAL * 1024) / 256, 256, 0, stream>>>(qkvb);

  // attention per (bt, head)
  attn_kernel<<<BT * NUM_HEADS, 256, 0, stream>>>(qkvb, ob);

  // out = O @ w_proj^T + b   (M=16384, N=1024, K=1024), fp32 output
  gemm_bf16_nt<true><<<dim3(D_MODEL / 128, M_TOTAL / 128), 128, 0, stream>>>(
      ob, wprojb, nullptr, out, b_proj, M_TOTAL, D_MODEL, D_MODEL);
}
